// LSTMCell_3728031613360
// MI455X (gfx1250) — matrix operand order repaired and checked by reading, not compiled, not run on hardware
//
#include <hip/hip_runtime.h>
#include <stdint.h>

// ---------------- CDNA5 WMMA fused LSTM cell (pipelined) ----------------
// pre = [x | h] @ [Wx ; Wh] + b   (K = 1024, N = 4 gates x 512)
// bf16 WMMA, f32 accum. Weights pre-converted to bf16 in d_ws when it fits.

typedef __attribute__((ext_vector_type(16))) __bf16 v16bf;
typedef __attribute__((ext_vector_type(8)))  float  v8f;

#define Bdim 32768
#define Ddim 512
#define Hdim 512
#define BM   128
#define BN   64
#define KT   32
#define LDA  40                      // bf16 elems per A row in LDS (80 B)
#define LDB  72                      // bf16 elems per B row in LDS (144 B)
#define SA_BYTES (BM * LDA * 2)      // 10240
#define SB_GATE  (KT * LDB * 2)      // 4608
#define SB_BYTES (4 * SB_GATE)       // 18432
#define WMAT_ELEMS (Ddim * Hdim)     // 262144 per weight matrix
#define NSTEPS 32                    // 2 phases x 16 K-steps

#if defined(__has_builtin)
#  if __has_builtin(__builtin_amdgcn_cvt_pk_bf16_f32)
#    define HAVE_CVT_PK_BF16 1
#  endif
#  if __has_builtin(__builtin_amdgcn_rcpf)
#    define HAVE_RCPF 1
#  endif
#endif

__device__ __forceinline__ uint32_t pk_bf16(float a, float b) {
#ifdef HAVE_CVT_PK_BF16
  typedef __attribute__((ext_vector_type(2))) __bf16 v2bf;
  v2bf r = __builtin_amdgcn_cvt_pk_bf16_f32(a, b);
  return __builtin_bit_cast(uint32_t, r);
#else
  // round-to-nearest-even f32 -> bf16, packed pair
  uint32_t ua = __float_as_uint(a), ub = __float_as_uint(b);
  ua += 0x7FFFu + ((ua >> 16) & 1u);
  ub += 0x7FFFu + ((ub >> 16) & 1u);
  return (ua >> 16) | (ub & 0xFFFF0000u);
#endif
}

__device__ __forceinline__ float fast_rcp(float x) {
#ifdef HAVE_RCPF
  return __builtin_amdgcn_rcpf(x);
#else
  return 1.0f / x;
#endif
}

__device__ __forceinline__ float sigmoidf_fast(float x) {
  return fast_rcp(1.0f + __expf(-x));
}

union Frag { uint4 q[2]; v16bf v; };

// ---------- pre-pass: convert the 8 weight matrices to bf16 in d_ws ----------
// layout: [phase][gate] each WMAT_ELEMS bf16; phase0 = w_i*, phase1 = w_h*
__global__ __launch_bounds__(256)
void cvt_weights_kernel(const float* __restrict__ s0, const float* __restrict__ s1,
                        const float* __restrict__ s2, const float* __restrict__ s3,
                        const float* __restrict__ s4, const float* __restrict__ s5,
                        const float* __restrict__ s6, const float* __restrict__ s7,
                        uint32_t* __restrict__ dst) {
  const float* S[8] = { s0, s1, s2, s3, s4, s5, s6, s7 };
  const float* src = S[blockIdx.y];
  uint32_t* d = dst + (size_t)blockIdx.y * (WMAT_ELEMS / 2);
  const int base = blockIdx.x * 256 + threadIdx.x;     // 32768 threads per matrix
  #pragma unroll
  for (int i = 0; i < 4; ++i) {
    const int p = base + i * 32768;                    // pair index, < 131072
    const float2 f = *(const float2*)(src + 2 * (size_t)p);
    d[p] = pk_bf16(f.x, f.y);
  }
}

// ---------------- main fused kernel ----------------
template <bool WB16>
__global__ __launch_bounds__(256, 1)
void lstm_wmma_kernel(const float* __restrict__ xp, const float* __restrict__ hp,
                      const float* __restrict__ cp,
                      const unsigned short* __restrict__ wq,   // bf16 weights (WB16)
                      const float* __restrict__ w_ii, const float* __restrict__ w_hi,
                      const float* __restrict__ b_ii, const float* __restrict__ b_hi,
                      const float* __restrict__ w_if, const float* __restrict__ w_hf,
                      const float* __restrict__ b_if, const float* __restrict__ b_hf,
                      const float* __restrict__ w_ig, const float* __restrict__ w_hg,
                      const float* __restrict__ b_ig, const float* __restrict__ b_hg,
                      const float* __restrict__ w_io, const float* __restrict__ w_ho,
                      const float* __restrict__ b_io, const float* __restrict__ b_ho,
                      float* __restrict__ out)
{
  __shared__ __align__(16) unsigned char smem[SA_BYTES + SB_BYTES];
  uint32_t*       sA32 = (uint32_t*)smem;
  unsigned short* sB16 = (unsigned short*)(smem + SA_BYTES);

  const int tid  = threadIdx.x;
  const int lane = tid & 31;                 // wave32
  const int wave = tid >> 5;                 // 8 waves: 4 (M) x 2 (N)
  const int m0   = blockIdx.x * BM;
  const int n0   = blockIdx.y * BN;
  const int mBase = (wave & 3) * 32;
  const int nBase = (wave >> 2) * 32;

  v8f acc[4][2][2] = {};                     // [gate][mt][nt]

  // staging assignments
  const int a_kq = (tid & 7) * 4;            // A: float4 per row chunk
  const int a_row = tid >> 3;
  const int b_n8 = (tid & 7) * 8;            // B bf16: uint4 (8 bf16) per (k,gate)
  const int b_kr = tid >> 3;                 // 0..31
  const int b_n2 = (tid & 31) * 2;           // B f32 fallback: float2
  const int b_kb = tid >> 5;

  // in-flight staging registers (loaded one step ahead of the ds_store)
  float4 ra[4];
  uint4  rb16[4];
  float2 rbf[16];

  auto issue_loads = [&](int step) {
    const int phase = step >> 4;
    const int k0 = (step & 15) * KT;
    const float* Ap = phase ? hp : xp;
    #pragma unroll
    for (int rr = 0; rr < 4; ++rr)
      ra[rr] = *(const float4*)(Ap + (size_t)(m0 + a_row + rr * 32) * Ddim + k0 + a_kq);
    if (WB16) {
      const unsigned short* wb = wq + (size_t)phase * 4 * WMAT_ELEMS;
      #pragma unroll
      for (int g = 0; g < 4; ++g)
        rb16[g] = *(const uint4*)(wb + (size_t)g * WMAT_ELEMS
                                  + (size_t)(k0 + b_kr) * Hdim + n0 + b_n8);
    } else {
      const float* W[4] = { phase ? w_hi : w_ii, phase ? w_hf : w_if,
                            phase ? w_hg : w_ig, phase ? w_ho : w_io };
      #pragma unroll
      for (int g = 0; g < 4; ++g)
        #pragma unroll
        for (int kk = 0; kk < 4; ++kk)
          rbf[g * 4 + kk] = *(const float2*)(W[g] + (size_t)(k0 + b_kb + kk * 8) * Hdim
                                             + n0 + b_n2);
    }
  };

  auto store_lds = [&]() {
    #pragma unroll
    for (int rr = 0; rr < 4; ++rr) {
      uint2 p; p.x = pk_bf16(ra[rr].x, ra[rr].y); p.y = pk_bf16(ra[rr].z, ra[rr].w);
      *(uint2*)(sA32 + (a_row + rr * 32) * (LDA / 2) + (a_kq >> 1)) = p;
    }
    if (WB16) {
      #pragma unroll
      for (int g = 0; g < 4; ++g)
        *(uint4*)(sB16 + g * (SB_GATE / 2) + b_kr * LDB + b_n8) = rb16[g];
    } else {
      #pragma unroll
      for (int g = 0; g < 4; ++g)
        #pragma unroll
        for (int kk = 0; kk < 4; ++kk) {
          const int k = b_kb + kk * 8;
          *(uint32_t*)(sB16 + g * (SB_GATE / 2) + k * LDB + b_n2) =
              pk_bf16(rbf[g * 4 + kk].x, rbf[g * 4 + kk].y);
        }
    }
  };

  issue_loads(0);

  for (int step = 0; step < NSTEPS; ++step) {
    __syncthreads();            // previous step's readers done with LDS
    store_lds();
    __syncthreads();            // tile visible
    if (step + 1 < NSTEPS)
      issue_loads(step + 1);    // overlap next tile's global latency with WMMA

    // ---- fragments + WMMA ----
    // A frag (16x32): lane<16 -> M=lane, K 0..7 & 16..23; lane>=16 -> K 8..15 & 24..31
    Frag a[2];
    {
      const int hsel = lane >> 4;
      #pragma unroll
      for (int mt = 0; mt < 2; ++mt) {
        const unsigned char* ab =
            smem + (size_t)(mBase + mt * 16 + (lane & 15)) * (LDA * 2) + hsel * 16;
        a[mt].q[0] = *(const uint4*)ab;
        a[mt].q[1] = *(const uint4*)(ab + 32);
      }
    }
    #pragma unroll
    for (int g = 0; g < 4; ++g) {
      // B frag (32x16): lane%16 = N col; lane<16 -> K 0..7 & 16..23; lane>=16 -> K 8..15 & 24..31
      Frag b[2];
      #pragma unroll
      for (int nt = 0; nt < 2; ++nt) {
        const unsigned short* bc = sB16 + g * (SB_GATE / 2) + nBase + nt * 16 + (lane & 15);
        const int kb = (lane >> 4) * 8;
        #pragma unroll
        for (int s = 0; s < 8; ++s) {
          b[nt].v[s]     = __builtin_bit_cast(__bf16, bc[(kb + s) * LDB]);
          b[nt].v[8 + s] = __builtin_bit_cast(__bf16, bc[(16 + kb + s) * LDB]);
        }
      }
      #pragma unroll
      for (int mt = 0; mt < 2; ++mt)
        #pragma unroll
        for (int nt = 0; nt < 2; ++nt)
          acc[g][mt][nt] = __builtin_amdgcn_wmma_f32_16x16x32_bf16(
              false, a[mt].v, false, b[nt].v,
              (short)0, acc[g][mt][nt], false, false);
    }
  }

  // ---- fused epilogue: bias + gates + state update ----
  const int hsel = lane >> 4;
  const int lcol = lane & 15;
  #pragma unroll
  for (int nt = 0; nt < 2; ++nt) {
    const int col = n0 + nBase + nt * 16 + lcol;
    const float bI = b_ii[col] + b_hi[col];
    const float bF = b_if[col] + b_hf[col];
    const float bG = b_ig[col] + b_hg[col];
    const float bO = b_io[col] + b_ho[col];
    #pragma unroll
    for (int mt = 0; mt < 2; ++mt) {
      #pragma unroll
      for (int r = 0; r < 8; ++r) {
        const int row = m0 + mBase + mt * 16 + r + hsel * 8;
        const size_t idx = (size_t)row * Hdim + col;
        const float iv = sigmoidf_fast(acc[0][mt][nt][r] + bI);
        const float fv = sigmoidf_fast(acc[1][mt][nt][r] + bF);
        const float gv = sigmoidf_fast(acc[2][mt][nt][r] + bG);  // ref: sigmoid for g
        const float ov = sigmoidf_fast(acc[3][mt][nt][r] + bO);
        const float cn = fv * cp[idx] + iv * gv;
        // tanh via exp2-based expf + rcp: tanh(x) = 1 - 2/(e^{2x}+1)
        const float e = __expf(2.0f * cn);
        const float th = 1.0f - 2.0f * fast_rcp(e + 1.0f);
        out[idx] = ov * th;                          // h'
        out[(size_t)Bdim * Hdim + idx] = cn;         // c'
      }
    }
  }
}

extern "C" void kernel_launch(void* const* d_in, const int* in_sizes, int n_in,
                              void* d_out, int out_size, void* d_ws, size_t ws_size,
                              hipStream_t stream) {
  const float* x    = (const float*)d_in[0];
  const float* h    = (const float*)d_in[1];
  const float* c    = (const float*)d_in[2];
  const float* w_ii = (const float*)d_in[3];
  const float* w_hi = (const float*)d_in[4];
  const float* b_ii = (const float*)d_in[5];
  const float* b_hi = (const float*)d_in[6];
  const float* w_if = (const float*)d_in[7];
  const float* w_hf = (const float*)d_in[8];
  const float* b_if = (const float*)d_in[9];
  const float* b_hf = (const float*)d_in[10];
  const float* w_ig = (const float*)d_in[11];
  const float* w_hg = (const float*)d_in[12];
  const float* b_ig = (const float*)d_in[13];
  const float* b_hg = (const float*)d_in[14];
  const float* w_io = (const float*)d_in[15];
  const float* w_ho = (const float*)d_in[16];
  const float* b_io = (const float*)d_in[17];
  const float* b_ho = (const float*)d_in[18];
  float* out = (float*)d_out;

  const size_t WQ_BYTES = 8ull * WMAT_ELEMS * 2ull;   // 4 MiB of bf16 weights
  dim3 grid(Bdim / BM, Hdim / BN);                    // 256 x 8
  dim3 block(256);

  if (d_ws != nullptr && ws_size >= WQ_BYTES) {
    // one-shot (per launch) weight conversion; order: w_i{i,f,g,o}, w_h{i,f,g,o}
    cvt_weights_kernel<<<dim3(128, 8), block, 0, stream>>>(
        w_ii, w_if, w_ig, w_io, w_hi, w_hf, w_hg, w_ho, (uint32_t*)d_ws);
    lstm_wmma_kernel<true><<<grid, block, 0, stream>>>(
        x, h, c, (const unsigned short*)d_ws,
        w_ii, w_hi, b_ii, b_hi, w_if, w_hf, b_if, b_hf,
        w_ig, w_hg, b_ig, b_hg, w_io, w_ho, b_io, b_ho, out);
  } else {
    lstm_wmma_kernel<false><<<grid, block, 0, stream>>>(
        x, h, c, nullptr,
        w_ii, w_hi, b_ii, b_hi, w_if, w_hf, b_if, b_hf,
        w_ig, w_hg, b_ig, b_hg, w_io, w_ho, b_io, b_ho, out);
  }
}
